// BaseComm_29214367547980
// MI455X (gfx1250) — matrix operand order repaired and checked by reading, not compiled, not run on hardware
//
#include <hip/hip_runtime.h>

typedef __attribute__((ext_vector_type(16))) __bf16 v16bf;
typedef __attribute__((ext_vector_type(8)))  __bf16 v8bf;
typedef __attribute__((ext_vector_type(8)))  float  v8f;
typedef __attribute__((ext_vector_type(4)))  float  v4f;

#define HID  128
#define MSGD 128
#define TWOH 256

// ---------------------------------------------------------------------------
// CDNA5 async global->LDS copy (ASYNCcnt-tracked), per CDNA5 ISA.
// Each active lane moves 16B from its global address to its LDS offset.
// ---------------------------------------------------------------------------
__device__ __forceinline__ void async_g2l_b128(unsigned lds_off, const void* gptr) {
  asm volatile("global_load_async_to_lds_b128 %0, %1, off"
               :: "v"(lds_off), "v"((unsigned long long)(size_t)gptr)
               : "memory");
}
__device__ __forceinline__ void wait_async0() {
  asm volatile("s_wait_asynccnt 0" ::: "memory");
}

// ---------------------------------------------------------------------------
// Fragment loaders from PRE-SPLIT bf16 tables (no conversion VALU in hot loops)
// ---------------------------------------------------------------------------
// A-fragment (16x32 bf16): lane L holds row L%16; element e holds
// k = (e/8)*16 + (L/16)*8 + (e%8) -> two contiguous 8-bf16 (16B) runs.
__device__ __forceinline__ v16bf load_frag_a(const __bf16* __restrict__ row, int half) {
  v8bf a = *reinterpret_cast<const v8bf*>(row + half * 8);
  v8bf b = *reinterpret_cast<const v8bf*>(row + 16 + half * 8);
  return __builtin_shufflevector(a, b, 0, 1, 2, 3, 4, 5, 6, 7,
                                 8, 9, 10, 11, 12, 13, 14, 15);
}

// B-fragment (32x16 bf16): lane n holds column n%16; element e holds
// k = (e/8)*16 + (n/16)*8 + (e%8) -> two contiguous 8-bf16 (16B) runs.
__device__ __forceinline__ v16bf load_frag_b(const __bf16* __restrict__ row, int half) {
  v8bf a = *reinterpret_cast<const v8bf*>(row + half * 8);
  v8bf b = *reinterpret_cast<const v8bf*>(row + 16 + half * 8);
  return __builtin_shufflevector(a, b, 0, 1, 2, 3, 4, 5, 6, 7,
                                 8, 9, 10, 11, 12, 13, 14, 15);
}

// On-the-fly split for c = msum * inv (only place conversion remains; done
// once per 16-node tile and reused across all column tiles / gates).
__device__ __forceinline__ void build_c_frag(const float* __restrict__ row, int half,
                                             float scale, v16bf& hi, v16bf& lo) {
  const v4f* p0 = reinterpret_cast<const v4f*>(row + half * 8);
  const v4f* p1 = reinterpret_cast<const v4f*>(row + 16 + half * 8);
  v4f q0 = p0[0], q1 = p0[1], q2 = p1[0], q3 = p1[1];
  float f[16] = {q0[0], q0[1], q0[2], q0[3], q1[0], q1[1], q1[2], q1[3],
                 q2[0], q2[1], q2[2], q2[3], q3[0], q3[1], q3[2], q3[3]};
#pragma unroll
  for (int e = 0; e < 16; ++e) {
    float v = f[e] * scale;
    __bf16 hv = (__bf16)v;
    hi[e] = hv;
    lo[e] = (__bf16)(v - (float)hv);
  }
}

// Error-compensated bf16 product: hi*hi + hi*lo + lo*hi (fp32 accumulate).
__device__ __forceinline__ v8f wmma3(v16bf ah, v16bf al, v16bf bh, v16bf bl, v8f acc) {
  acc = __builtin_amdgcn_wmma_f32_16x16x32_bf16(false, ah, false, bh, (short)0, acc, false, false);
  acc = __builtin_amdgcn_wmma_f32_16x16x32_bf16(false, ah, false, bl, (short)0, acc, false, false);
  acc = __builtin_amdgcn_wmma_f32_16x16x32_bf16(false, al, false, bh, (short)0, acc, false, false);
  return acc;
}

// ---------------------------------------------------------------------------
// Prepass kernels
// ---------------------------------------------------------------------------
__global__ void zero_ws_kernel(float* __restrict__ p, long long n) {
  long long i = (long long)blockIdx.x * blockDim.x + threadIdx.x;
  long long stride = (long long)gridDim.x * blockDim.x;
  for (; i < n; i += stride) p[i] = 0.0f;
}

// f32 -> (bf16 hi, bf16 lo) split tables.
__global__ void split_kernel(const float* __restrict__ in, __bf16* __restrict__ hi,
                             __bf16* __restrict__ lo, long long n) {
  long long i = (long long)blockIdx.x * blockDim.x + threadIdx.x;
  long long stride = (long long)gridDim.x * blockDim.x;
  for (; i < n; i += stride) {
    float v = in[i];
    __bf16 hv = (__bf16)v;
    hi[i] = hv;
    lo[i] = (__bf16)(v - (float)hv);
  }
}

// ---------------------------------------------------------------------------
// Kernel: fused edge message GEMM + scatter (persistent waves).
// W_msg hi/lo (128KB) is async-staged once per workgroup into LDS.
// Per edge tile (16 edges x all 128 msg dims): ks-outer / jt-inner with 8
// independent accumulator chains -> scheduler bursts WMMAs and pipelines the
// B ds_loads. A 32-bit LDS *offset* is laundered through a volatile asm no-op
// inside the tile loop: the B-load addresses stay loop-variant (LICM cannot
// hoist/spill them) while pointer provenance still proves LDS -> ds_load.
// Bounds are wave-uniform -> EXEC stays all-ones for WMMA.
// ---------------------------------------------------------------------------
extern __shared__ __bf16 sW[];   // [0..32767] = W_msg hi, [32768..65535] = lo

__global__ __launch_bounds__(256) void edge_msg_kernel(
    const __bf16* __restrict__ xhi, const __bf16* __restrict__ xlo,
    const __bf16* __restrict__ hhi, const __bf16* __restrict__ hlo,
    const int* __restrict__ src, const int* __restrict__ dst,
    const __bf16* __restrict__ Wh, const __bf16* __restrict__ Wl,
    const float* __restrict__ bmsg,
    float* __restrict__ msum, float* __restrict__ deg, int E, int ntiles) {
  const int NW = MSGD * TWOH;                // 32768 bf16 per table
  // --- async-stage split W_msg into LDS (each lane moves 16B per issue) ---
  for (int i = threadIdx.x * 8; i < NW; i += 256 * 8) {
    async_g2l_b128((unsigned)(size_t)(void*)(sW + i), Wh + i);
    async_g2l_b128((unsigned)(size_t)(void*)(sW + NW + i), Wl + i);
  }
  wait_async0();
  __syncthreads();

  int wid = threadIdx.x >> 5;
  int lane = threadIdx.x & 31;
  int half = lane >> 4;
  int mrow = lane & 15;

#pragma unroll 1
  for (int tile = blockIdx.x * 8 + wid; tile < ntiles; tile += gridDim.x * 8) {
    // Opaque zero offset, fresh each iteration: keeps the B loads inside the
    // loop (no LICM spill-hoisting) without losing the LDS address space.
    unsigned lz = 0;
    asm volatile("" : "+v"(lz));
    const __bf16* sWp = sW + lz;

    int e0 = tile * 16;
    int eA = e0 + mrow; if (eA >= E) eA = E - 1;
    size_t abase = (size_t)src[eA] * HID;

    v8f acc[8];
#pragma unroll
    for (int jt = 0; jt < 8; ++jt) acc[jt] = (v8f){0, 0, 0, 0, 0, 0, 0, 0};

#pragma unroll
    for (int ks = 0; ks < 8; ++ks) {         // K = 256 in steps of 32
      const __bf16* rh = (ks < 4) ? (xhi + abase + ks * 32) : (hhi + abase + (ks - 4) * 32);
      const __bf16* rl = (ks < 4) ? (xlo + abase + ks * 32) : (hlo + abase + (ks - 4) * 32);
      v16bf ah = load_frag_a(rh, half);
      v16bf al = load_frag_a(rl, half);
#pragma unroll
      for (int jt = 0; jt < 8; ++jt) {       // N = 128 in tiles of 16
        int woff = (jt * 16 + mrow) * TWOH + ks * 32;
        v16bf bh = load_frag_b(sWp + woff, half);       // ds_load from LDS
        v16bf bl = load_frag_b(sWp + NW + woff, half);
        acc[jt] = wmma3(ah, al, bh, bl, acc[jt]);
      }
    }

    int dstM[8];
#pragma unroll
    for (int r = 0; r < 8; ++r) {
      int e = e0 + r + 8 * half;             // D row M = r + 8*half
      dstM[r] = (e < E) ? dst[e] : -1;
    }
#pragma unroll
    for (int jt = 0; jt < 8; ++jt) {
      int j = jt * 16 + mrow;                // D col
      float bj = bmsg[j];
#pragma unroll
      for (int r = 0; r < 8; ++r) {
        if (dstM[r] >= 0)
          atomicAdd(&msum[(size_t)dstM[r] * MSGD + j], acc[jt][r] + bj);
      }
    }
    if (half == 0) {                         // lanes 0..15 -> one edge each
      int e = e0 + mrow;
      if (e < E) atomicAdd(&deg[dst[e]], 1.0f);
    }
  }
}

// ---------------------------------------------------------------------------
// Kernel: fused GRU cell. One wave = 16 nodes.
// c = msum/max(deg,1) is split once per tile (registers), reused for all jt.
// r,z gates accumulate inp@W_ih^T and h@W_hh^T jointly; i_n / h_n kept apart.
// (Split W_ih+W_hh is 576KB > LDS; it stays in L2 which is plenty here.)
// ---------------------------------------------------------------------------
__global__ __launch_bounds__(256) void gru_kernel(
    const __bf16* __restrict__ xhi, const __bf16* __restrict__ xlo,
    const __bf16* __restrict__ hhi, const __bf16* __restrict__ hlo,
    const float* __restrict__ h,
    const float* __restrict__ msum, const float* __restrict__ deg,
    const __bf16* __restrict__ Wih_h, const __bf16* __restrict__ Wih_l,
    const __bf16* __restrict__ Whh_h, const __bf16* __restrict__ Whh_l,
    const float* __restrict__ bih, const float* __restrict__ bhh,
    float* __restrict__ out, int N) {
  int wid = threadIdx.x >> 5;
  int lane = threadIdx.x & 31;
  int tile = blockIdx.x * 8 + wid;
  int n0 = tile * 16;
  if (n0 >= N) return;                       // wave-uniform exit

  int half = lane >> 4;
  int mrow = lane & 15;
  int nodeA = n0 + mrow; if (nodeA >= N) nodeA = N - 1;
  size_t abase = (size_t)nodeA * HID;
  const float* crow = msum + (size_t)nodeA * MSGD;
  float inv = 1.0f / fmaxf(deg[nodeA], 1.0f);

  __builtin_prefetch(Wih_h, 0, 1);           // global_prefetch_b8

  // Split c once per tile; reuse across all 8 column tiles and 3 gates.
  v16bf ch[4], cl[4];
#pragma unroll
  for (int ks = 0; ks < 4; ++ks) build_c_frag(crow + ks * 32, half, inv, ch[ks], cl[ks]);

  const v8f zacc = {0, 0, 0, 0, 0, 0, 0, 0};

#pragma unroll 1
  for (int jt = 0; jt < 8; ++jt) {           // 16-wide slices of the H=128 output
    int jc = jt * 16 + mrow;                 // this lane's output column
    v8f aR = zacc, aZ = zacc, aIN = zacc, aHN = zacc;

    // inp = [x | c] against W_ih (row stride 256), gates r, z, n
#pragma unroll
    for (int ks = 0; ks < 8; ++ks) {
      v16bf ah, al;
      if (ks < 4) {
        ah = load_frag_a(xhi + abase + ks * 32, half);
        al = load_frag_a(xlo + abase + ks * 32, half);
      } else {
        ah = ch[ks - 4];
        al = cl[ks - 4];
      }
      size_t o0 = (size_t)jc * TWOH + ks * 32;
      size_t o1 = (size_t)(HID + jc) * TWOH + ks * 32;
      size_t o2 = (size_t)(2 * HID + jc) * TWOH + ks * 32;
      aR  = wmma3(ah, al, load_frag_b(Wih_h + o0, half), load_frag_b(Wih_l + o0, half), aR);
      aZ  = wmma3(ah, al, load_frag_b(Wih_h + o1, half), load_frag_b(Wih_l + o1, half), aZ);
      aIN = wmma3(ah, al, load_frag_b(Wih_h + o2, half), load_frag_b(Wih_l + o2, half), aIN);
    }
    // h against W_hh (row stride 128), gates r, z, n
#pragma unroll
    for (int ks = 0; ks < 4; ++ks) {
      v16bf ah = load_frag_a(hhi + abase + ks * 32, half);
      v16bf al = load_frag_a(hlo + abase + ks * 32, half);
      size_t o0 = (size_t)jc * HID + ks * 32;
      size_t o1 = (size_t)(HID + jc) * HID + ks * 32;
      size_t o2 = (size_t)(2 * HID + jc) * HID + ks * 32;
      aR  = wmma3(ah, al, load_frag_b(Whh_h + o0, half), load_frag_b(Whh_l + o0, half), aR);
      aZ  = wmma3(ah, al, load_frag_b(Whh_h + o1, half), load_frag_b(Whh_l + o1, half), aZ);
      aHN = wmma3(ah, al, load_frag_b(Whh_h + o2, half), load_frag_b(Whh_l + o2, half), aHN);
    }

    float bR  = bih[jc] + bhh[jc];
    float bZ  = bih[HID + jc] + bhh[HID + jc];
    float bN  = bih[2 * HID + jc];
    float bHN = bhh[2 * HID + jc];
#pragma unroll
    for (int r = 0; r < 8; ++r) {
      int node = n0 + r + 8 * half;          // D row M = r + 8*half
      if (node < N) {
        float rg = 1.0f / (1.0f + __expf(-(aR[r] + bR)));
        float zg = 1.0f / (1.0f + __expf(-(aZ[r] + bZ)));
        float ng = tanhf(aIN[r] + bN + rg * (aHN[r] + bHN));
        float hv = h[(size_t)node * HID + jc];
        out[(size_t)node * HID + jc] = (1.0f - zg) * ng + zg * hv;
      }
    }
  }
}

// ---------------------------------------------------------------------------
// Host entry
// ---------------------------------------------------------------------------
static inline size_t align_up(size_t v, size_t a) { return (v + a - 1) & ~(a - 1); }

extern "C" void kernel_launch(void* const* d_in, const int* in_sizes, int n_in,
                              void* d_out, int out_size, void* d_ws, size_t ws_size,
                              hipStream_t stream) {
  const float* x    = (const float*)d_in[0];
  const float* h    = (const float*)d_in[1];
  const int*   src  = (const int*)d_in[2];
  const int*   dst  = (const int*)d_in[3];
  const float* Wmsg = (const float*)d_in[4];
  const float* bmsg = (const float*)d_in[5];
  const float* Wih  = (const float*)d_in[6];
  const float* Whh  = (const float*)d_in[7];
  const float* bih  = (const float*)d_in[8];
  const float* bhh  = (const float*)d_in[9];
  float* out = (float*)d_out;

  int N = in_sizes[0] / HID;
  int E = in_sizes[2];
  long long nfeat  = (long long)N * HID;     // x/h element count
  long long nWmsg  = in_sizes[4];            // 128*256
  long long nWih   = in_sizes[6];            // 384*256
  long long nWhh   = in_sizes[7];            // 384*128

  // --- carve workspace (all sections 64B aligned) ---
  char* cur = (char*)d_ws;
  auto carve = [&](size_t bytes) { char* p = cur; cur += align_up(bytes, 64); return p; };
  float*  msum   = (float*)carve((size_t)N * MSGD * 4);
  float*  deg    = (float*)carve((size_t)N * 4);
  __bf16* xhi    = (__bf16*)carve((size_t)nfeat * 2);
  __bf16* xlo    = (__bf16*)carve((size_t)nfeat * 2);
  __bf16* hhi    = (__bf16*)carve((size_t)nfeat * 2);
  __bf16* hlo    = (__bf16*)carve((size_t)nfeat * 2);
  __bf16* wmsg_h = (__bf16*)carve((size_t)nWmsg * 2);
  __bf16* wmsg_l = (__bf16*)carve((size_t)nWmsg * 2);
  __bf16* wih_h  = (__bf16*)carve((size_t)nWih * 2);
  __bf16* wih_l  = (__bf16*)carve((size_t)nWih * 2);
  __bf16* whh_h  = (__bf16*)carve((size_t)nWhh * 2);
  __bf16* whh_l  = (__bf16*)carve((size_t)nWhh * 2);

  // --- prepass: zero accumulators, pre-split operands to bf16 hi/lo ---
  zero_ws_kernel<<<2048, 256, 0, stream>>>(msum, (long long)N * MSGD);
  zero_ws_kernel<<<256, 256, 0, stream>>>(deg, (long long)N);
  split_kernel<<<2048, 256, 0, stream>>>(x, xhi, xlo, nfeat);
  split_kernel<<<2048, 256, 0, stream>>>(h, hhi, hlo, nfeat);
  split_kernel<<<128, 256, 0, stream>>>(Wmsg, wmsg_h, wmsg_l, nWmsg);
  split_kernel<<<384, 256, 0, stream>>>(Wih, wih_h, wih_l, nWih);
  split_kernel<<<192, 256, 0, stream>>>(Whh, whh_h, whh_l, nWhh);

  // --- fused edge GEMM + scatter (persistent, LDS-staged weights) ---
  int etiles = (E + 15) / 16;
  int eblocks = (etiles + 7) / 8;
  if (eblocks > 1024) eblocks = 1024;        // persistent waves, ~6 tiles each
  size_t ldsBytes = (size_t)MSGD * TWOH * 2 * sizeof(__bf16);  // 128KB
  edge_msg_kernel<<<eblocks, 256, ldsBytes, stream>>>(
      xhi, xlo, hhi, hlo, src, dst, wmsg_h, wmsg_l, bmsg, msum, deg, E, etiles);

  // --- fused GRU update ---
  int ntiles = (N + 15) / 16;
  gru_kernel<<<(ntiles + 7) / 8, 256, 0, stream>>>(
      xhi, xlo, hhi, hlo, h, msum, deg, wih_h, wih_l, whh_h, whh_l,
      bih, bhh, out, N);
}
